// GAT_33878702030947
// MI455X (gfx1250) — compile-verified, matrix operand order checked by reading, not run on hardware
//
#include <hip/hip_runtime.h>
#include <math.h>

typedef __attribute__((ext_vector_type(4)))  float    v4f;
typedef __attribute__((ext_vector_type(8)))  float    v8f;
typedef __attribute__((ext_vector_type(16))) _Float16 v16h;

#define N_TOTAL 100000
#define IN_DIM  128
#define HID     64
#define NH      8
#define OUTC    (NH * HID)      // 512
#define CLASSN  40
#define NBATCH  10000
#define DEG     32
#define ALPHA   0.2f

#define MT        (N_TOTAL / 16)   // 6250 row tiles
#define NT        (OUTC / 16)      // 32 col tiles
#define KCHUNKS   (IN_DIM / 32)    // 4 chunks of K=32
#define NGROUPS   4                // 4 ntile groups of 8 (one wave each)
#define MT_PER_BLK 8

// K index inside a 16-bit WMMA fragment (ISA 7.12.2, 16-bit A/B layout):
// half hh of lane l -> K = 32*chunk + hh + (hh>=8 ? 8 : 0) + (l>=16 ? 8 : 0)

// ---------------------------------------------------------------------------
// Pack W (8 x 128 x 64 f32) into per-lane v16h B fragments, hi/lo f16 planes.
// Bfrag index: ((ntile*4 + chunk)*32 + lane)*16 + hh
// ---------------------------------------------------------------------------
__global__ __launch_bounds__(256) void gat_pack_b(
    const float* __restrict__ W, _Float16* __restrict__ Bhi,
    _Float16* __restrict__ Blo)
{
    int idx = blockIdx.x * 256 + threadIdx.x;      // 32*4*32*16 = 65536
    int hh   = idx & 15;
    int lane = (idx >> 4) & 31;
    int c    = (idx >> 9) & 3;
    int nt   = idx >> 11;
    int K    = c * 32 + hh + (hh >= 8 ? 8 : 0) + ((lane >> 4) ? 8 : 0);
    int bcol = nt * 16 + (lane & 15);
    int wh = bcol >> 6, wo = bcol & 63;
    float x = W[(size_t)wh * (IN_DIM * HID) + (size_t)K * HID + wo];
    _Float16 hi = (_Float16)x;
    _Float16 lo = (_Float16)(x - (float)hi);
    Bhi[idx] = hi;
    Blo[idx] = lo;
}

// ---------------------------------------------------------------------------
// Kernel 1: h_all = gf (100000x128) @ B (128x512) via split-precision f16
// WMMA: acc += Ahi*Bhi + Ahi*Blo + Alo*Bhi (f32 accumulate, ~21-bit mantissa).
// Block = 8 waves; wave w owns ntile = group*8 + w with all B fragments held
// in VGPRs. Block loops over 8 mtiles, staging each 16x128 A tile in LDS as
// hi/lo fragments (converted f32->f16 cooperatively).
// ---------------------------------------------------------------------------
__global__ __launch_bounds__(256) void gat_project(
    const float* __restrict__ gf, const _Float16* __restrict__ Bhi,
    const _Float16* __restrict__ Blo, float* __restrict__ h_all)
{
    __shared__ _Float16 afh[KCHUNKS * 32 * 16] __attribute__((aligned(32)));
    __shared__ _Float16 afl[KCHUNKS * 32 * 16] __attribute__((aligned(32)));

    int t    = threadIdx.x;
    int wave = t >> 5;
    int lane = t & 31;
    int group  = blockIdx.x & (NGROUPS - 1);
    int mblock = blockIdx.x >> 2;
    int ntile  = group * 8 + wave;

    // B fragments: resident in registers for the whole kernel (64 VGPRs).
    v16h bhi[KCHUNKS], blo[KCHUNKS];
    #pragma unroll
    for (int c = 0; c < KCHUNKS; ++c) {
        size_t fo = ((size_t)(ntile * KCHUNKS + c) * 32 + lane) * 16;
        bhi[c] = *(const v16h*)(Bhi + fo);
        blo[c] = *(const v16h*)(Blo + fo);
    }

    // A-staging role of this thread: 8 consecutive halves of one fragment.
    int s_hh0 = (t & 1) * 8;          // 0 or 8
    int s_l   = (t >> 1) & 31;        // fragment lane
    int s_c   = t >> 6;               // chunk
    int s_row = s_l & 15;
    int s_k   = s_c * 32 + (s_hh0 ? 16 : 0) + ((s_l >> 4) ? 8 : 0);
    int s_pos = t * 8;                // ((s_c*32 + s_l)*16 + s_hh0)

    int bcol = ntile * 16 + (lane & 15);
    int mrow_off = (lane >> 4) * 8;

    for (int mi = 0; mi < MT_PER_BLK; ++mi) {
        int mtile = mblock * MT_PER_BLK + mi;
        if (mtile >= MT) break;                    // uniform across block

        __syncthreads();   // previous iteration's fragment reads done
        {
            const float* src = gf + ((size_t)mtile * 16 + s_row) * IN_DIM + s_k;
            v4f x0 = *(const v4f*)(src);
            v4f x1 = *(const v4f*)(src + 4);
            #pragma unroll
            for (int j = 0; j < 4; ++j) {
                _Float16 h0 = (_Float16)x0[j];
                afh[s_pos + j] = h0;
                afl[s_pos + j] = (_Float16)(x0[j] - (float)h0);
                _Float16 h1 = (_Float16)x1[j];
                afh[s_pos + 4 + j] = h1;
                afl[s_pos + 4 + j] = (_Float16)(x1[j] - (float)h1);
            }
        }
        __syncthreads();

        v8f acc = {};
        #pragma unroll
        for (int c = 0; c < KCHUNKS; ++c) {
            v16h ah = *(const v16h*)(afh + (c * 32 + lane) * 16);
            v16h al = *(const v16h*)(afl + (c * 32 + lane) * 16);
            acc = __builtin_amdgcn_wmma_f32_16x16x32_f16(
                false, ah, false, bhi[c], (short)0, acc, false, false);
            acc = __builtin_amdgcn_wmma_f32_16x16x32_f16(
                false, ah, false, blo[c], (short)0, acc, false, false);
            acc = __builtin_amdgcn_wmma_f32_16x16x32_f16(
                false, al, false, bhi[c], (short)0, acc, false, false);
        }

        float* out = h_all + ((size_t)mtile * 16 + mrow_off) * OUTC + bcol;
        #pragma unroll
        for (int v = 0; v < 8; ++v)
            out[(size_t)v * OUTC] = acc[v];
    }
}

// ---------------------------------------------------------------------------
// Kernel 2: s1[n,h] = h_all[n,h,:] . a[h,0:64]; s2[n,h] = . a[h,64:128]
// ---------------------------------------------------------------------------
__global__ __launch_bounds__(256) void gat_dots(
    const float* __restrict__ h_all, const float* __restrict__ a,
    float* __restrict__ s1, float* __restrict__ s2)
{
    int idx = blockIdx.x * 256 + threadIdx.x;
    if (idx >= N_TOTAL * NH) return;
    int n = idx >> 3, h = idx & 7;
    const float* hv = h_all + (size_t)n * OUTC + h * HID;
    const float* a1 = a + h * (2 * HID);
    const float* a2 = a1 + HID;
    float acc1 = 0.f, acc2 = 0.f;
    #pragma unroll
    for (int o = 0; o < HID; o += 4) {
        v4f hvv = *(const v4f*)(hv + o);
        v4f a1v = *(const v4f*)(a1 + o);
        v4f a2v = *(const v4f*)(a2 + o);
        acc1 += hvv.x * a1v.x + hvv.y * a1v.y + hvv.z * a1v.z + hvv.w * a1v.w;
        acc2 += hvv.x * a2v.x + hvv.y * a2v.y + hvv.z * a2v.z + hvv.w * a2v.w;
    }
    s1[idx] = acc1;
    s2[idx] = acc2;
}

// ---------------------------------------------------------------------------
// Kernel 3: per-node softmax over 32 neighbors, aggregation, ELU, head-mean,
// classifier. One 256-thread block per node; bound by the 32 x 2KB gathers
// (h_all is ~L2-resident at 192MB).
// ---------------------------------------------------------------------------
__global__ __launch_bounds__(256) void gat_attend(
    const float* __restrict__ h_all, const float* __restrict__ s1,
    const float* __restrict__ s2, const int* __restrict__ nodes,
    const int* __restrict__ neighbors, const float* __restrict__ Wc,
    const float* __restrict__ bc, float* __restrict__ newf_out,
    float* __restrict__ logit_out)
{
    __shared__ int   nb[DEG];
    __shared__ float attn[DEG][NH];
    __shared__ float aggsm[NH][HID];
    __shared__ float newf[HID];

    int b = blockIdx.x;
    int t = threadIdx.x;

    if (t < DEG) nb[t] = neighbors[b * DEG + t];
    __syncthreads();

    {
        int k = t >> 3, h = t & 7;
        int node = nodes[b];
        float e = s1[node * NH + h] + s2[nb[k] * NH + h];
        attn[k][h] = (e > 0.f) ? e : ALPHA * e;
    }
    __syncthreads();

    if (t < NH) {
        float mx = -INFINITY;
        for (int k = 0; k < DEG; ++k) mx = fmaxf(mx, attn[k][t]);
        float sum = 0.f;
        for (int k = 0; k < DEG; ++k) {
            float v = __expf(attn[k][t] - mx);
            attn[k][t] = v;
            sum += v;
        }
        float inv = 1.f / sum;
        for (int k = 0; k < DEG; ++k) attn[k][t] *= inv;
    }
    __syncthreads();

    int d  = t & 63;
    int h0 = t >> 6;                  // 0..3 ; thread covers heads h0, h0+4
    float agg0 = 0.f, agg1 = 0.f;
    for (int k = 0; k < DEG; ++k) {
        const float* hv = h_all + (size_t)nb[k] * OUTC;
        agg0 += attn[k][h0]     * hv[h0 * HID + d];
        agg1 += attn[k][h0 + 4] * hv[(h0 + 4) * HID + d];
    }
    agg0 = (agg0 > 0.f) ? agg0 : (__expf(agg0) - 1.f);
    agg1 = (agg1 > 0.f) ? agg1 : (__expf(agg1) - 1.f);
    aggsm[h0][d]     = agg0;
    aggsm[h0 + 4][d] = agg1;
    __syncthreads();

    if (t < HID) {
        float s = 0.f;
        #pragma unroll
        for (int h = 0; h < NH; ++h) s += aggsm[h][t];
        s *= (1.f / NH);
        newf[t] = s;
        newf_out[(size_t)b * HID + t] = s;
    }
    __syncthreads();

    if (t < CLASSN) {
        float acc = bc[t];
        for (int dd = 0; dd < HID; ++dd)
            acc += newf[dd] * Wc[dd * CLASSN + t];
        logit_out[(size_t)b * CLASSN + t] = acc;
    }
}

extern "C" void kernel_launch(void* const* d_in, const int* in_sizes, int n_in,
                              void* d_out, int out_size, void* d_ws, size_t ws_size,
                              hipStream_t stream) {
    (void)in_sizes; (void)n_in; (void)out_size; (void)ws_size;
    const float* gf        = (const float*)d_in[0];
    const int*   nodes     = (const int*)d_in[1];
    const int*   neighbors = (const int*)d_in[2];
    const float* W         = (const float*)d_in[3];
    const float* a         = (const float*)d_in[4];
    const float* Wc        = (const float*)d_in[5];
    const float* bc        = (const float*)d_in[6];

    float* out       = (float*)d_out;
    float* newf_out  = out;                               // 10000 x 64
    float* logit_out = out + (size_t)NBATCH * HID;        // 10000 x 40

    // Workspace layout (all 32B-aligned offsets)
    float*    h_all = (float*)d_ws;                          // 100000 x 512 f32
    float*    s1    = h_all + (size_t)N_TOTAL * OUTC;        // 100000 x 8
    float*    s2    = s1 + (size_t)N_TOTAL * NH;             // 100000 x 8
    _Float16* Bhi   = (_Float16*)(s2 + (size_t)N_TOTAL * NH); // 65536 f16
    _Float16* Blo   = Bhi + (size_t)NT * KCHUNKS * 32 * 16;   // 65536 f16

    gat_pack_b<<<(NT * KCHUNKS * 32 * 16) / 256, 256, 0, stream>>>(W, Bhi, Blo);

    int mblocks = (MT + MT_PER_BLK - 1) / MT_PER_BLK;     // 782
    gat_project<<<mblocks * NGROUPS, 256, 0, stream>>>(gf, Bhi, Blo, h_all);

    int dots = N_TOTAL * NH;
    gat_dots<<<(dots + 255) / 256, 256, 0, stream>>>(h_all, a, s1, s2);

    gat_attend<<<NBATCH, 256, 0, stream>>>(h_all, s1, s2, nodes, neighbors,
                                           Wc, bc, newf_out, logit_out);
}
